// MultiHeadCausalSelfAttention_81870666596905
// MI455X (gfx1250) — matrix operand order repaired and checked by reading, not compiled, not run on hardware
//
#include <hip/hip_runtime.h>
#include <hip/hip_bf16.h>
#include <stdint.h>

// ---------------------------------------------------------------------------
// MI455X (gfx1250) multi-head causal self-attention, bf16 WMMA pipeline.
// cvt->bf16 ; QKV GEMM (WMMA + TDM tensor_load_to_lds for A tiles) ;
// flash attention (async global->LDS K/Q tiles, WMMA QK^T + online softmax +
// WMMA PV) ; out-proj GEMM (WMMA, fp32 epilogue).
// ---------------------------------------------------------------------------

typedef __attribute__((ext_vector_type(16))) __bf16 v16bf;
typedef __attribute__((ext_vector_type(8)))  float  v8f;
typedef unsigned int u32x4 __attribute__((ext_vector_type(4)));
typedef int i32x4 __attribute__((ext_vector_type(4)));
typedef int i32x8 __attribute__((ext_vector_type(8)));

struct U4x2 { uint4 a, b; };  // 32 bytes == v16bf

__device__ __forceinline__ unsigned short f2bf(float f) {
  unsigned int u = __float_as_uint(f);
  unsigned int r = u + 0x7FFFu + ((u >> 16) & 1u);  // round-to-nearest-even
  return (unsigned short)(r >> 16);
}

__device__ __forceinline__ v8f vzero8() {
  v8f z;
#pragma unroll
  for (int i = 0; i < 8; i++) z[i] = 0.0f;
  return z;
}

// Per-lane async global->LDS copy of 16 bytes (GLOBAL_LOAD_ASYNC_TO_LDS_B128,
// GVS mode: 64-bit SGPR base + 32-bit VGPR byte offset). Tracked by ASYNCcnt.
__device__ __forceinline__ void async_copy_b128(unsigned lds_byte_addr,
                                                unsigned goff_bytes,
                                                const void* sbase) {
  asm volatile("global_load_async_to_lds_b128 %0, %1, %2"
               :
               : "v"(lds_byte_addr), "v"(goff_bytes),
                 "s"((unsigned long long)(uintptr_t)sbase)
               : "memory");
}

__device__ __forceinline__ void async_wait0() {
  asm volatile("s_wait_asynccnt 0x0" ::: "memory");
}

// TDM: DMA a [rows x 32] bf16 tile (row stride = ld elements) into LDS.
// Issued uniformly (wave-level); caller waits on TENSORcnt.
__device__ __forceinline__ void tdm_load_tile_b16(unsigned lds_byte_addr,
                                                  const void* gtile, int ld,
                                                  int rows, int tensor_rows) {
  unsigned long long ga = (unsigned long long)(uintptr_t)gtile;
  u32x4 g0;
  g0.x = 1u;                                           // count=1, user mode
  g0.y = lds_byte_addr;                                // lds_addr
  g0.z = (unsigned)(ga & 0xFFFFFFFFu);                 // global_addr[31:0]
  g0.w = (unsigned)((ga >> 32) & 0x1FFFFFFu) | (2u << 30);  // addr[56:32],type=2
  i32x8 g1;
  g1[0] = (int)(1u << 16);                             // data_size=1 (2 bytes)
  g1[1] = (int)(((unsigned)ld & 0xFFFFu) << 16);       // tensor_dim0[15:0]
  g1[2] = (int)((((unsigned)ld >> 16) & 0xFFFFu) |
                (((unsigned)tensor_rows & 0xFFFFu) << 16));  // dim0 hi, dim1 lo
  g1[3] = (int)((((unsigned)tensor_rows >> 16) & 0xFFFFu) |
                (32u << 16));                          // dim1 hi, tile_dim0=32
  g1[4] = rows;                                        // tile_dim1, tile_dim2=0
  g1[5] = ld;                                          // tensor_dim0_stride lo
  g1[6] = 0;                                           // stride hi, dim1_stride lo
  g1[7] = 0;
  i32x4 gz = {0, 0, 0, 0};
#if defined(__clang_major__) && (__clang_major__ >= 23)
  i32x8 gz8 = {0, 0, 0, 0, 0, 0, 0, 0};
  __builtin_amdgcn_tensor_load_to_lds(g0, g1, gz, gz, gz8, 0);
#else
  __builtin_amdgcn_tensor_load_to_lds(g0, g1, gz, gz, 0);
#endif
}

// A-fragment (16x32 bf16, MxK) per ISA: lane m=lane&15, kh=lane>>4;
// elems 0..7 -> K = kh*8+e ; elems 8..15 -> K = 16+kh*8+e. Row-major LDS.
__device__ __forceinline__ v16bf load_a_frag(const unsigned short* lds, int mbase,
                                             int kbase, int ldk, int lane) {
  int m = lane & 15, kh = lane >> 4;
  const unsigned short* p = lds + (mbase + m) * ldk + kbase + kh * 8;
  U4x2 r;
  r.a = *(const uint4*)(p);
  r.b = *(const uint4*)(p + 16);
  return __builtin_bit_cast(v16bf, r);
}

// B-fragment (32x16 bf16, KxN): lane n=lane&15, kh=lane>>4; elems 0..7 -> K=kh*8+e ; elems 8..15 -> K=16+kh*8+e.
// LDS indexed [n][k] with stride ldk so K-values per lane are contiguous.
__device__ __forceinline__ v16bf load_b_frag(const unsigned short* lds, int nbase,
                                             int kbase, int ldk, int lane) {
  int n = lane & 15, kh = lane >> 4;
  const unsigned short* p = lds + (nbase + n) * ldk + kbase + kh * 8;
  U4x2 r;
  r.a = *(const uint4*)(p);
  r.b = *(const uint4*)(p + 16);
  return __builtin_bit_cast(v16bf, r);
}

__device__ __forceinline__ v8f wmma_bf16(v16bf a, v16bf b, v8f c) {
  return __builtin_amdgcn_wmma_f32_16x16x32_bf16(false, a, false, b, (short)0, c,
                                                 false, false);
}

// ---------------------------------------------------------------------------
__global__ void cvt_f32_bf16(const float* __restrict__ src,
                             unsigned short* __restrict__ dst, int n) {
  int i = blockIdx.x * blockDim.x + threadIdx.x;
  int stride = gridDim.x * blockDim.x;
  for (; i < n; i += stride) dst[i] = f2bf(src[i]);
}

// ---------------------------------------------------------------------------
// 128x128x32-tiled bf16 GEMM, 256 threads = 8 waves; wave -> 32x64 (2x4 tiles).
// A tile staged via TDM (tensor_load_to_lds), B tile manually transposed.
// MODE 0: C scattered as bf16 into Q/K/V [B,H,S,Hd]; Q pre-scaled by 1/sqrt(Hd).
// MODE 1: C written as fp32 row-major.
// ---------------------------------------------------------------------------
template <int MODE>
__global__ __launch_bounds__(256) void gemm128(
    const unsigned short* __restrict__ A, const unsigned short* __restrict__ B,
    unsigned short* __restrict__ Qo, unsigned short* __restrict__ Ko,
    unsigned short* __restrict__ Vo, float* __restrict__ Cout, int M, int N, int K) {
  __shared__ alignas(16) unsigned short As[128 * 32];   // row-major [m][k]
  __shared__ alignas(16) unsigned short Bt[128 * 32];   // [n][k]
  const int tid = threadIdx.x;
  const int lane = tid & 31, wid = tid >> 5;
  const int waveM = wid >> 1, waveN = wid & 1;
  const int m0 = blockIdx.y * 128, n0 = blockIdx.x * 128;
  const unsigned lds_a = (unsigned)(uintptr_t)&As[0];

  v8f acc[2][4];
#pragma unroll
  for (int i = 0; i < 2; i++)
#pragma unroll
    for (int j = 0; j < 4; j++) acc[i][j] = vzero8();

  for (int k0 = 0; k0 < K; k0 += 32) {
    __syncthreads();
    // A tile 128x32 via Tensor Data Mover (one descriptor, issued by wave 0)
    if (wid == 0) {
      tdm_load_tile_b16(lds_a, A + (size_t)m0 * K + k0, K, 128, M);
    }
    // B tile: 32x128, stored transposed [n][k]
#pragma unroll 2
    for (int c = tid; c < 512; c += 256) {
      int r = c >> 4, col8 = (c & 15) * 8;
      uint4 d = *(const uint4*)&B[(size_t)(k0 + r) * N + n0 + col8];
      const unsigned short* s = (const unsigned short*)&d;
#pragma unroll
      for (int e = 0; e < 8; e++) Bt[(col8 + e) * 32 + r] = s[e];
    }
    if (k0 + 32 < K) {  // hint next A tile toward L2 (global_prefetch_b8)
      __builtin_prefetch(&A[(size_t)(m0 + (tid >> 1)) * K + k0 + 32], 0, 1);
    }
    if (wid == 0) {
      __builtin_amdgcn_s_wait_tensorcnt(0);
    }
    __syncthreads();

    v16bf af[2], bf[4];
#pragma unroll
    for (int mt = 0; mt < 2; mt++)
      af[mt] = load_a_frag(As, waveM * 32 + mt * 16, 0, 32, lane);
#pragma unroll
    for (int nt = 0; nt < 4; nt++)
      bf[nt] = load_b_frag(Bt, waveN * 64 + nt * 16, 0, 32, lane);
#pragma unroll
    for (int mt = 0; mt < 2; mt++)
#pragma unroll
      for (int nt = 0; nt < 4; nt++)
        acc[mt][nt] = wmma_bf16(af[mt], bf[nt], acc[mt][nt]);
  }

  const int rbase = m0 + waveM * 32 + (lane >> 4) * 8;
  const int cbase = n0 + waveN * 64 + (lane & 15);
  const float qscale = 0.08838834764831845f;  // 1/sqrt(128)
#pragma unroll
  for (int mt = 0; mt < 2; mt++) {
#pragma unroll
    for (int nt = 0; nt < 4; nt++) {
#pragma unroll
      for (int v = 0; v < 8; v++) {
        int row = rbase + mt * 16 + v;
        int col = cbase + nt * 16;
        float val = acc[mt][nt][v];
        if (MODE == 0) {
          int mat = col >> 11;            // 0:Q 1:K 2:V
          int d = col & 2047;
          int h = d >> 7, hd = d & 127;
          int b = row >> 11, s = row & 2047;
          size_t off = ((size_t)(b * 16 + h) * 2048 + s) * 128 + hd;
          unsigned short bfv = f2bf(mat == 0 ? val * qscale : val);
          if (mat == 0)      Qo[off] = bfv;
          else if (mat == 1) Ko[off] = bfv;
          else               Vo[off] = bfv;
        } else {
          Cout[(size_t)row * N + col] = val;
        }
      }
    }
  }
}

// ---------------------------------------------------------------------------
// Flash attention: block = 128 threads (4 waves), one (b,h, 64-row qtile).
// Q/K tiles staged with global_load_async_to_lds_b128; V transposed manually.
// ---------------------------------------------------------------------------
__global__ __launch_bounds__(128) void flash_attn(
    const unsigned short* __restrict__ Qg, const unsigned short* __restrict__ Kg,
    const unsigned short* __restrict__ Vg, unsigned short* __restrict__ Og) {
  __shared__ alignas(16) unsigned short Qs[64 * 128];  // [q][hd]
  __shared__ alignas(16) unsigned short Ks[64 * 128];  // [key][hd] (B-frag: n=key)
  __shared__ alignas(16) unsigned short Vt[128 * 64];  // [hd][key] (B-frag: n=hd)
  __shared__ alignas(16) unsigned short Ps[64 * 64];   // [q][key] bf16 probs

  const int tid = threadIdx.x;
  const int lane = tid & 31, wid = tid >> 5;
  const int qt = blockIdx.x;   // 64-row query tile index
  const int bh = blockIdx.y;   // b*16 + h
  const size_t base = (size_t)bh * 2048 * 128;
  const int qbase = qt * 64;
  const unsigned short* Qbase = Qg + base;
  const unsigned short* Kbase = Kg + base;

  // Q tile 64x128 via async global->LDS
  for (int c = tid; c < 1024; c += 128) {
    int r = c >> 4, col8 = (c & 15) * 8;
    async_copy_b128((unsigned)(uintptr_t)&Qs[r * 128 + col8],
                    (unsigned)((((qbase + r) * 128) + col8) * 2), Qbase);
  }
  async_wait0();
  __syncthreads();

  v16bf qf[4];
#pragma unroll
  for (int ks = 0; ks < 4; ks++) qf[ks] = load_a_frag(Qs, wid * 16, ks * 32, 128, lane);

  v8f o[8];
#pragma unroll
  for (int i = 0; i < 8; i++) o[i] = vzero8();
  float mrow[8], lrow[8];
#pragma unroll
  for (int v = 0; v < 8; v++) { mrow[v] = -INFINITY; lrow[v] = 0.0f; }

  const int myrow0 = wid * 16 + (lane >> 4) * 8;  // local q row base (this lane)
  const int ncol = lane & 15;                     // N index within a 16-wide tile

  for (int jt = 0; jt <= qt; jt++) {
    const int kb = jt * 64;
    __syncthreads();
    for (int c = tid; c < 1024; c += 128) {  // K async row-major, V transposed
      int r = c >> 4, col8 = (c & 15) * 8;
      async_copy_b128((unsigned)(uintptr_t)&Ks[r * 128 + col8],
                      (unsigned)((((kb + r) * 128) + col8) * 2), Kbase);
      uint4 d = *(const uint4*)&Vg[base + (size_t)(kb + r) * 128 + col8];
      const unsigned short* s = (const unsigned short*)&d;
#pragma unroll
      for (int e = 0; e < 8; e++) Vt[(col8 + e) * 64 + r] = s[e];
    }
    async_wait0();
    __syncthreads();

    // S(16x64 per wave) = Q * K^T  (scale already folded into Q)
    v8f st[4];
#pragma unroll
    for (int nt = 0; nt < 4; nt++) st[nt] = vzero8();
#pragma unroll
    for (int ks = 0; ks < 4; ks++) {
#pragma unroll
      for (int nt = 0; nt < 4; nt++) {
        v16bf kf = load_b_frag(Ks, nt * 16, ks * 32, 128, lane);
        st[nt] = wmma_bf16(qf[ks], kf, st[nt]);
      }
    }
    if (jt == qt) {  // causal mask only on the diagonal tile
#pragma unroll
      for (int nt = 0; nt < 4; nt++)
#pragma unroll
        for (int v = 0; v < 8; v++) {
          int cg = kb + nt * 16 + ncol;
          int rg = qbase + myrow0 + v;
          if (cg > rg) st[nt][v] = -INFINITY;
        }
    }

    // online softmax: row stats via shfl-xor across the 16-lane N group
    float alpha[8];
#pragma unroll
    for (int v = 0; v < 8; v++) {
      float m = fmaxf(fmaxf(st[0][v], st[1][v]), fmaxf(st[2][v], st[3][v]));
#pragma unroll
      for (int d = 1; d < 16; d <<= 1) m = fmaxf(m, __shfl_xor(m, d, 32));
      float mnew = fmaxf(mrow[v], m);
      alpha[v] = __expf(mrow[v] - mnew);
      mrow[v] = mnew;
    }
    float rsum[8];
#pragma unroll
    for (int v = 0; v < 8; v++) rsum[v] = 0.0f;
#pragma unroll
    for (int nt = 0; nt < 4; nt++) {
#pragma unroll
      for (int v = 0; v < 8; v++) {
        float p = __expf(st[nt][v] - mrow[v]);
        rsum[v] += p;
        Ps[(myrow0 + v) * 64 + nt * 16 + ncol] = f2bf(p);
      }
    }
#pragma unroll
    for (int v = 0; v < 8; v++) {
      float s = rsum[v];
#pragma unroll
      for (int d = 1; d < 16; d <<= 1) s += __shfl_xor(s, d, 32);
      lrow[v] = lrow[v] * alpha[v] + s;
    }
#pragma unroll
    for (int nt = 0; nt < 8; nt++)
#pragma unroll
      for (int v = 0; v < 8; v++) o[nt][v] *= alpha[v];
    __syncthreads();  // Ps (C-layout) -> A-fragment reshape via LDS

    // O += P * V
#pragma unroll
    for (int ks = 0; ks < 2; ks++) {
      v16bf pf = load_a_frag(&Ps[wid * 16 * 64], 0, ks * 32, 64, lane);
#pragma unroll
      for (int nt = 0; nt < 8; nt++) {
        v16bf vf = load_b_frag(Vt, nt * 16, ks * 32, 64, lane);
        o[nt] = wmma_bf16(pf, vf, o[nt]);
      }
    }
  }

  const int b = bh >> 4, h = bh & 15;
#pragma unroll
  for (int nt = 0; nt < 8; nt++) {
#pragma unroll
    for (int v = 0; v < 8; v++) {
      int rg = qbase + myrow0 + v;
      int col = h * 128 + nt * 16 + ncol;
      float val = o[nt][v] / lrow[v];
      Og[((size_t)(b * 2048 + rg)) * 2048 + col] = f2bf(val);
    }
  }
}

// ---------------------------------------------------------------------------
extern "C" void kernel_launch(void* const* d_in, const int* in_sizes, int n_in,
                              void* d_out, int out_size, void* d_ws, size_t ws_size,
                              hipStream_t stream) {
  const float* x    = (const float*)d_in[0];
  const float* Wqkv = (const float*)d_in[1];
  const float* Wo   = (const float*)d_in[2];
  float* out = (float*)d_out;
  (void)in_sizes; (void)n_in; (void)out_size; (void)ws_size;

  const int D = 2048;
  const size_t MS = 4096;  // B*S

  char* ws = (char*)d_ws;
  size_t off = 0;
  auto alloc = [&](size_t bytes) {
    char* p = ws + off;
    off += (bytes + 255) & ~(size_t)255;
    return p;
  };
  unsigned short* xb  = (unsigned short*)alloc(MS * D * 2);            // 16 MB
  unsigned short* wqb = (unsigned short*)alloc((size_t)D * 3 * D * 2); // 24 MB
  unsigned short* wob = (unsigned short*)alloc((size_t)D * D * 2);     //  8 MB
  unsigned short* Qb  = (unsigned short*)alloc(MS * D * 2);            // 16 MB
  unsigned short* Kb  = (unsigned short*)alloc(MS * D * 2);
  unsigned short* Vb  = (unsigned short*)alloc(MS * D * 2);
  unsigned short* Ab  = (unsigned short*)alloc(MS * D * 2);

  cvt_f32_bf16<<<2048, 256, 0, stream>>>(x, xb, (int)(MS * D));
  cvt_f32_bf16<<<2048, 256, 0, stream>>>(Wqkv, wqb, D * 3 * D);
  cvt_f32_bf16<<<2048, 256, 0, stream>>>(Wo, wob, D * D);

  // QKV projection: (4096 x 2048) x (2048 x 6144)
  gemm128<0><<<dim3(48, 32), 256, 0, stream>>>(xb, wqb, Qb, Kb, Vb, nullptr,
                                               4096, 6144, 2048);
  // Flash attention: 32 query tiles x 32 (b,h) pairs
  flash_attn<<<dim3(32, 32), 128, 0, stream>>>(Qb, Kb, Vb, Ab);
  // Output projection: (4096 x 2048) x (2048 x 2048) -> fp32
  gemm128<1><<<dim3(16, 32), 256, 0, stream>>>(Ab, wob, nullptr, nullptr, nullptr,
                                               out, 4096, 2048, 2048);
}
